// LSTMModel_35390530519090
// MI455X (gfx1250) — compile-verified, matrix operand order checked by reading, not run on hardware
//
#include <hip/hip_runtime.h>
#include <hip/hip_bf16.h>

// Problem dimensions (compile-time constants from the reference)
#define B_    16384
#define T_    336
#define DIN_  7
#define H_    4
#define DOUT_ 7
#define K_    (T_ * H_)      // 1344 = 42 * 32
#define N_    (T_ * DOUT_)   // 2352 = 49 * 48

// GEMM tiling
#define KSTEP_    32                              // K per WMMA / LDS stage
#define NSTEPS_   (K_ / KSTEP_)                   // 42 (even)
#define MTILE_    256                             // block M tile
#define NTILE_    48                              // block N tile
#define A_STG_    (MTILE_ * KSTEP_ * 2)           // 16384 B
#define B_STG_    (NTILE_ * KSTEP_ * 2)           // 3072 B
#define STG_      (A_STG_ + B_STG_)               // 19456 B per stage
#define NTHR_     384                             // 12 waves: 4 (M) x 3 (N)

typedef __attribute__((ext_vector_type(16))) __bf16 v16bf;
typedef __attribute__((ext_vector_type(8)))  float  v8f;
typedef __attribute__((ext_vector_type(4)))  unsigned int u32x4;

// ---------------------------------------------------------------------------
// CDNA5 async-copy + counter helpers
// ---------------------------------------------------------------------------
__device__ __forceinline__ void async_copy_b128(unsigned lds_off, const void* g) {
    // ASYNCcnt-tracked DMA: 16B per lane, global -> LDS, no VGPR data.
    asm volatile("global_load_async_to_lds_b128 %0, %1, off"
                 :: "v"(lds_off), "v"((unsigned long long)g)
                 : "memory");
}

#if defined(__has_builtin)
#if __has_builtin(__builtin_amdgcn_s_wait_asynccnt)
#define WAIT_ASYNC(n) __builtin_amdgcn_s_wait_asynccnt(n)
#endif
#endif
#ifndef WAIT_ASYNC
#define WAIT_ASYNC(n) asm volatile("s_wait_asynccnt %0" :: "i"(n) : "memory")
#endif

// ---------------------------------------------------------------------------
// Fast activations (one v_exp_f32 each)
// ---------------------------------------------------------------------------
__device__ __forceinline__ float sigm_(float x) {
    return 1.0f / (1.0f + __expf(-x));
}
__device__ __forceinline__ float tanh_(float x) {
    return 2.0f / (1.0f + __expf(-2.0f * x)) - 1.0f;   // tanh = 2*sig(2x)-1
}

// ---------------------------------------------------------------------------
// Kernel 1: convert w_lin (fp32 [N_, K_]) -> bf16, row-major unchanged.
// ---------------------------------------------------------------------------
__global__ __launch_bounds__(256) void wlin_to_bf16_kernel(
    const float* __restrict__ w, __bf16* __restrict__ wb, int n)
{
    int i = blockIdx.x * 256 + threadIdx.x;
    if (i < n) wb[i] = (__bf16)w[i];
}

// ---------------------------------------------------------------------------
// Kernel 2: fused two-layer LSTM. One thread per batch element runs the
// full T=336 recurrence for both layers; weights in LDS (broadcast reads).
// Layer-1 hidden states written as bf16 in GEMM A layout: A[b, t*4+j].
// ---------------------------------------------------------------------------
__global__ __launch_bounds__(256) void lstm2_fused_kernel(
    const float* __restrict__ x,
    const float* __restrict__ w_ih0, const float* __restrict__ w_hh0,
    const float* __restrict__ b_ih0, const float* __restrict__ b_hh0,
    const float* __restrict__ w_ih1, const float* __restrict__ w_hh1,
    const float* __restrict__ b_ih1, const float* __restrict__ b_hh1,
    __bf16* __restrict__ a_out)     // [B_, K_] bf16
{
    __shared__ float s_wih0[16 * DIN_];
    __shared__ float s_whh0[16 * H_];
    __shared__ float s_wih1[16 * H_];
    __shared__ float s_whh1[16 * H_];
    __shared__ float s_b0[16];
    __shared__ float s_b1[16];

    const int tid = threadIdx.x;
    if (tid < 16 * DIN_) s_wih0[tid] = w_ih0[tid];
    if (tid < 16 * H_) {
        s_whh0[tid] = w_hh0[tid];
        s_wih1[tid] = w_ih1[tid];
        s_whh1[tid] = w_hh1[tid];
    }
    if (tid < 16) {
        s_b0[tid] = b_ih0[tid] + b_hh0[tid];
        s_b1[tid] = b_ih1[tid] + b_hh1[tid];
    }
    __syncthreads();

    const int b = blockIdx.x * 256 + tid;
    const float* __restrict__ xp = x + (long)b * T_ * DIN_;
    __bf16* __restrict__ op = a_out + (long)b * K_;

    float h0[H_] = {0.f, 0.f, 0.f, 0.f};
    float c0[H_] = {0.f, 0.f, 0.f, 0.f};
    float h1[H_] = {0.f, 0.f, 0.f, 0.f};
    float c1[H_] = {0.f, 0.f, 0.f, 0.f};

#pragma unroll 1
    for (int t = 0; t < T_; ++t) {
        if (t + 8 < T_) __builtin_prefetch(xp + (t + 8) * DIN_, 0, 0);

        float xi[DIN_];
#pragma unroll
        for (int d = 0; d < DIN_; ++d) xi[d] = xp[t * DIN_ + d];

        float g0[16];
#pragma unroll
        for (int g = 0; g < 16; ++g) {
            float a = s_b0[g];
#pragma unroll
            for (int d = 0; d < DIN_; ++d) a = fmaf(s_wih0[g * DIN_ + d], xi[d], a);
#pragma unroll
            for (int j = 0; j < H_; ++j)   a = fmaf(s_whh0[g * H_ + j], h0[j], a);
            g0[g] = a;
        }
#pragma unroll
        for (int j = 0; j < H_; ++j) {
            float ig = sigm_(g0[j]);
            float fg = sigm_(g0[4 + j]);
            float gg = tanh_(g0[8 + j]);
            float og = sigm_(g0[12 + j]);
            float cn = fg * c0[j] + ig * gg;
            c0[j] = cn;
            h0[j] = og * tanh_(cn);
        }

        float g1[16];
#pragma unroll
        for (int g = 0; g < 16; ++g) {
            float a = s_b1[g];
#pragma unroll
            for (int d = 0; d < H_; ++d) a = fmaf(s_wih1[g * H_ + d], h0[d], a);
#pragma unroll
            for (int j = 0; j < H_; ++j) a = fmaf(s_whh1[g * H_ + j], h1[j], a);
            g1[g] = a;
        }
#pragma unroll
        for (int j = 0; j < H_; ++j) {
            float ig = sigm_(g1[j]);
            float fg = sigm_(g1[4 + j]);
            float gg = tanh_(g1[8 + j]);
            float og = sigm_(g1[12 + j]);
            float cn = fg * c1[j] + ig * gg;
            c1[j] = cn;
            float hn = og * tanh_(cn);
            h1[j] = hn;
            op[t * H_ + j] = (__bf16)hn;
        }
    }
}

// ---------------------------------------------------------------------------
// Kernel 3: WMMA GEMM  C[M,N] = A[M,K](bf16) * W[N,K](bf16)^T + bias[N]
//
// 384 threads (12 waves, 4 M x 3 N), C tile 256x48, K staged 32 at a time
// into double-buffered LDS via global_load_async_to_lds_b128 (ASYNCcnt).
// Copy roles are per-wave with exact, wave-uniform counts:
//   waves 0-7 : A tile, 4 b128/thread (1024 total)
//   waves 8-9 : B tile, 3 b128/thread (192 total)
//   waves 10-11: no copies
// Compute reads fragments with ds_load_b128 (stage base is a compile-time
// constant thanks to a 2x unrolled pipeline) and issues 4 WMMA per wave
// per K-step.
// ---------------------------------------------------------------------------
__device__ __forceinline__ v16bf ld_frag(const char* p) {
    union { v16bf v; u32x4 q[2]; } f;
    f.q[0] = *(const u32x4*)p;
    f.q[1] = *(const u32x4*)(p + 32);
    return f.v;
}

__global__ __launch_bounds__(NTHR_) void wmma_linear_kernel(
    const __bf16* __restrict__ A,    // [M, K] bf16
    const __bf16* __restrict__ W,    // [N, K] bf16 (w_lin rows)
    const float*  __restrict__ bias, // [N]
    float* __restrict__ C)           // [M, N] fp32
{
    __shared__ __align__(16) char lds[2 * STG_];

    const int tid   = threadIdx.x;
    const int lane  = tid & 31;
    const int wave  = tid >> 5;          // 0..11
    const int l15   = lane & 15;
    const int lhalf = lane >> 4;         // 0 / 1
    const int mwave = wave & 3;          // 4 M sub-tiles of 64
    const int nwave = wave >> 2;         // 3 N sub-tiles of 16

    const int n0 = blockIdx.x * NTILE_;  // 49 tiles
    const int m0 = blockIdx.y * MTILE_;  // 64 tiles

    const unsigned lds_u = (unsigned)(size_t)lds;

    // ---- per-thread async-copy descriptors (wave-uniform counts) --------
    const char* gsrc[4];
    unsigned    loff[4];                 // LDS offset within a stage
#pragma unroll
    for (int jj = 0; jj < 4; ++jj) { gsrc[jj] = (const char*)A; loff[jj] = 0; }
    if (wave < 8) {
#pragma unroll
        for (int jj = 0; jj < 4; ++jj) {
            int i = tid + jj * 256;               // 0..1023
            int row = i >> 2, ch = i & 3;
            gsrc[jj] = (const char*)(A + (long)(m0 + row) * K_) + ch * 16;
            loff[jj] = (unsigned)(row * 64 + ch * 16);
        }
    } else if (wave < 10) {
        const int local = tid - 256;              // 0..63
#pragma unroll
        for (int jj = 0; jj < 3; ++jj) {
            int i = local + jj * 64;              // 0..191
            int row = i >> 2, ch = i & 3;
            gsrc[jj] = (const char*)(W + (long)(n0 + row) * K_) + ch * 16;
            loff[jj] = (unsigned)(A_STG_ + row * 64 + ch * 16);
        }
    }

    // ---- K-invariant LDS fragment offsets (ints, added to lds directly) -
    int aoff[4];
#pragma unroll
    for (int mi = 0; mi < 4; ++mi)
        aoff[mi] = (mwave * 64 + mi * 16 + l15) * 64 + lhalf * 16;
    const int boff = A_STG_ + (nwave * 16 + l15) * 64 + lhalf * 16;

    v8f acc0 = {}, acc1 = {}, acc2 = {}, acc3 = {};

    // ---- prologue: stage 0 in flight ------------------------------------
    if (wave < 8) {
#pragma unroll
        for (int jj = 0; jj < 4; ++jj) async_copy_b128(lds_u + loff[jj], gsrc[jj]);
    } else if (wave < 10) {
#pragma unroll
        for (int jj = 0; jj < 3; ++jj) async_copy_b128(lds_u + loff[jj], gsrc[jj]);
    }

    // One pipeline sub-step; CURBASE/NXTBASE are compile-time stage bases.
#define GEMM_SUBSTEP(CURBASE, NXTBASE, SV)                                     \
    do {                                                                       \
        if ((SV) + 1 < NSTEPS_) {                                              \
            const unsigned kb = (unsigned)((SV) + 1) * (KSTEP_ * 2);           \
            if (wave < 8) {                                                    \
                _Pragma("unroll")                                              \
                for (int jj = 0; jj < 4; ++jj)                                 \
                    async_copy_b128(lds_u + (NXTBASE) + loff[jj], gsrc[jj] + kb); \
                WAIT_ASYNC(4);  /* stage SV resident (in-order completion) */  \
            } else if (wave < 10) {                                            \
                _Pragma("unroll")                                              \
                for (int jj = 0; jj < 3; ++jj)                                 \
                    async_copy_b128(lds_u + (NXTBASE) + loff[jj], gsrc[jj] + kb); \
                WAIT_ASYNC(3);                                                 \
            }                                                                  \
        } else {                                                               \
            WAIT_ASYNC(0);                                                     \
        }                                                                      \
        __syncthreads();    /* stage SV visible to all waves */                \
        {                                                                      \
            v16bf fb  = ld_frag(lds + (CURBASE) + boff);                       \
            v16bf fa0 = ld_frag(lds + (CURBASE) + aoff[0]);                    \
            v16bf fa1 = ld_frag(lds + (CURBASE) + aoff[1]);                    \
            v16bf fa2 = ld_frag(lds + (CURBASE) + aoff[2]);                    \
            v16bf fa3 = ld_frag(lds + (CURBASE) + aoff[3]);                    \
            acc0 = __builtin_amdgcn_wmma_f32_16x16x32_bf16(                    \
                false, fa0, false, fb, (short)0, acc0, false, false);          \
            acc1 = __builtin_amdgcn_wmma_f32_16x16x32_bf16(                    \
                false, fa1, false, fb, (short)0, acc1, false, false);          \
            acc2 = __builtin_amdgcn_wmma_f32_16x16x32_bf16(                    \
                false, fa2, false, fb, (short)0, acc2, false, false);          \
            acc3 = __builtin_amdgcn_wmma_f32_16x16x32_bf16(                    \
                false, fa3, false, fb, (short)0, acc3, false, false);          \
        }                                                                      \
        __syncthreads();    /* reads of stage SV done before overwrite */      \
    } while (0)

#pragma unroll 1
    for (int s = 0; s < NSTEPS_; s += 2) {
        GEMM_SUBSTEP(0,     STG_,  s);        // compute buf0, stream into buf1
        GEMM_SUBSTEP(STG_,  0,     s + 1);    // compute buf1, stream into buf0
    }
#undef GEMM_SUBSTEP

    // ---- epilogue: add bias, store fp32 ---------------------------------
    const long col = n0 + nwave * 16 + l15;
    const float bv = bias[col];
    const int mbase = m0 + mwave * 64 + lhalf * 8;

#pragma unroll
    for (int r = 0; r < 8; ++r) {
        C[(long)(mbase +      r) * N_ + col] = acc0[r] + bv;
        C[(long)(mbase + 16 + r) * N_ + col] = acc1[r] + bv;
        C[(long)(mbase + 32 + r) * N_ + col] = acc2[r] + bv;
        C[(long)(mbase + 48 + r) * N_ + col] = acc3[r] + bv;
    }
}

// ---------------------------------------------------------------------------
// Host-side launcher
// ---------------------------------------------------------------------------
extern "C" void kernel_launch(void* const* d_in, const int* in_sizes, int n_in,
                              void* d_out, int out_size, void* d_ws, size_t ws_size,
                              hipStream_t stream) {
    const float* x     = (const float*)d_in[0];
    const float* w_ih0 = (const float*)d_in[1];
    const float* w_hh0 = (const float*)d_in[2];
    const float* b_ih0 = (const float*)d_in[3];
    const float* b_hh0 = (const float*)d_in[4];
    const float* w_ih1 = (const float*)d_in[5];
    const float* w_hh1 = (const float*)d_in[6];
    const float* b_ih1 = (const float*)d_in[7];
    const float* b_hh1 = (const float*)d_in[8];
    const float* w_lin = (const float*)d_in[9];
    const float* b_lin = (const float*)d_in[10];
    float* out = (float*)d_out;

    // Workspace: A bf16 [B_, K_] then W bf16 [N_, K_], both 256B-aligned.
    __bf16* Abf = (__bf16*)d_ws;
    __bf16* Wbf = (__bf16*)((char*)d_ws + (size_t)B_ * K_ * sizeof(__bf16));

    // 1) w_lin -> bf16
    {
        const int n = N_ * K_;
        wlin_to_bf16_kernel<<<(n + 255) / 256, 256, 0, stream>>>(w_lin, Wbf, n);
    }

    // 2) fused 2-layer LSTM -> A matrix (bf16, GEMM-ready layout)
    lstm2_fused_kernel<<<B_ / 256, 256, 0, stream>>>(
        x, w_ih0, w_hh0, b_ih0, b_hh0, w_ih1, w_hh1, b_ih1, b_hh1, Abf);

    // 3) async-LDS double-buffered WMMA GEMM + bias -> out
    {
        dim3 grid(N_ / NTILE_, B_ / MTILE_);   // 49 x 64
        wmma_linear_kernel<<<grid, NTHR_, 0, stream>>>(Abf, Wbf, b_lin, out);
    }
}
